// CrossAttention_16810501997343
// MI455X (gfx1250) — matrix operand order repaired and checked by reading, not compiled, not run on hardware
//
#include <hip/hip_runtime.h>

typedef __attribute__((ext_vector_type(16))) _Float16 v16h;
typedef __attribute__((ext_vector_type(8)))  float    v8f;
typedef __attribute__((ext_vector_type(4)))  int      v4i;
#define H16 _Float16

#if defined(__has_builtin)
#  if __has_builtin(__builtin_amdgcn_global_load_async_to_lds_b128)
#    define HAVE_ASYNC_LDS 1
#  else
#    define HAVE_ASYNC_LDS 0
#  endif
#  if __has_builtin(__builtin_amdgcn_s_wait_asynccnt)
#    define HAVE_WAIT_ASYNC 1
#  else
#    define HAVE_WAIT_ASYNC 0
#  endif
#else
#  define HAVE_ASYNC_LDS 0
#  define HAVE_WAIT_ASYNC 0
#endif

typedef __attribute__((address_space(1))) v4i gv4i_t;   // global b128
typedef __attribute__((address_space(3))) v4i lv4i_t;   // LDS b128

static __device__ __forceinline__ void async_copy_b128(const H16* gsrc, H16* ldst) {
#if HAVE_ASYNC_LDS
  __builtin_amdgcn_global_load_async_to_lds_b128(
      (gv4i_t*)gsrc, (lv4i_t*)ldst, 0, 0);
#else
  *(float4*)ldst = *(const float4*)gsrc;
#endif
}

static __device__ __forceinline__ void wait_async_lds() {
#if HAVE_ASYNC_LDS
#  if HAVE_WAIT_ASYNC
  __builtin_amdgcn_s_wait_asynccnt(0);
#  else
  asm volatile("s_wait_asynccnt 0x0" ::: "memory");
#  endif
#endif
}

static __device__ __forceinline__ v8f wmma_f16(const v16h& a, const v16h& b, const v8f& c) {
  // D = A(16x32 f16) * B(32x16 f16) + C(16x16 f32)
  return __builtin_amdgcn_wmma_f32_16x16x32_f16(false, a, false, b, (short)0, c, false, false);
}

union Frag16 { v16h h; unsigned int u[8]; };
union Half8  { float4 v; H16 h[8]; };

// ---------------------------------------------------------------- convert
__global__ __launch_bounds__(256) void cvt_f32_f16(const float* __restrict__ s,
                                                   H16* __restrict__ d, int n) {
  int i = (blockIdx.x * 256 + threadIdx.x) * 8;
  if (i >= n) return;
  float4 a = *(const float4*)(s + i);
  float4 b = *(const float4*)(s + i + 4);
  Half8 o;
  o.h[0] = (H16)a.x; o.h[1] = (H16)a.y; o.h[2] = (H16)a.z; o.h[3] = (H16)a.w;
  o.h[4] = (H16)b.x; o.h[5] = (H16)b.y; o.h[6] = (H16)b.z; o.h[7] = (H16)b.w;
  *(float4*)(d + i) = o.v;
}

// ---------------------------------------------------------------- GEMM
// C[M x Nmat] = A[M x K] * W[K x Nmat], f16 in, f32 accumulate.
// MODE 0: write f16 to head-layout [b,h,n,d]; cols >= split go to out1 (col-split).
// MODE 1: write f32 row-major with bias add.
// Block: 256 thr = 8 waves; tile 128x128, K-step 32; wave computes 32x64.
// A tile staged via GLOBAL_LOAD_ASYNC_TO_LDS (ASYNCcnt); B tile transposed manually.
template<int MODE>
__global__ __launch_bounds__(256) void gemm128(
    const H16* __restrict__ A, const H16* __restrict__ W,
    int M, int Nmat, int K,
    H16* __restrict__ out0, H16* __restrict__ out1, int split,
    float* __restrict__ outF, const float* __restrict__ bias) {
  __shared__ H16 As[128 * 40];   // [row][k], stride 40 (16B-aligned rows)
  __shared__ H16 Bs[128 * 40];   // transposed: Bt[n][k], stride 40
  const int t = threadIdx.x;
  const int lane = t & 31, wave = t >> 5;
  const int wm = wave >> 1, wn = wave & 1;
  const int g = lane >> 4, ln = lane & 15;
  const int row0 = blockIdx.y * 128, col0 = blockIdx.x * 128;

  const int sArow = t >> 1, sAcol = (t & 1) * 16;
  const int sBkp = t & 15, sBn = (t >> 4) * 8;

  v8f acc[2][4] = {};

  for (int kk = 0; kk < K; kk += 32) {
    // stage A tile 128x32: async DMA straight into LDS (no VGPR round trip)
    const H16* ap = A + (size_t)(row0 + sArow) * K + kk + sAcol;
    async_copy_b128(ap,     &As[sArow * 40 + sAcol]);
    async_copy_b128(ap + 8, &As[sArow * 40 + sAcol + 8]);
    // stage B tile 32x128 transposed into Bt[n][k] (k-pairs packed b32)
    Half8 b0, b1;
    const H16* wp = W + (size_t)(kk + 2 * sBkp) * Nmat + col0 + sBn;
    b0.v = *(const float4*)wp;
    b1.v = *(const float4*)(wp + Nmat);
#pragma unroll
    for (int j = 0; j < 8; j++) {
      unsigned int lo = *(const unsigned short*)&b0.h[j];
      unsigned int hi = *(const unsigned short*)&b1.h[j];
      *(unsigned int*)&Bs[(sBn + j) * 40 + 2 * sBkp] = lo | (hi << 16);
    }
    if (kk + 32 < K) {           // prefetch next K-step tiles into cache
      __builtin_prefetch(ap + 32, 0, 3);
      __builtin_prefetch(wp + 32 * Nmat, 0, 3);
    }
    wait_async_lds();
    __syncthreads();

    // A fragments (16x32): lane ln = row, VGPR v -> k = (v/4)*16 + g*8 + (v%4)*2
    Frag16 af[2];
#pragma unroll
    for (int tm = 0; tm < 2; tm++) {
      int r = wm * 32 + tm * 16 + ln;
#pragma unroll
      for (int v = 0; v < 8; v++) {
        int k0 = ((v >> 2) << 4) + (g << 3) + ((v & 3) << 1);
        af[tm].u[v] = *(const unsigned int*)&As[r * 40 + k0];
      }
    }
    // B fragments (32x16): lane ln = col, k = (v/4)*16 + g*8 + (v%4)*2
#pragma unroll
    for (int tn = 0; tn < 4; tn++) {
      Frag16 bf;
      int n = wn * 64 + tn * 16 + ln;
#pragma unroll
      for (int v = 0; v < 8; v++)
        bf.u[v] = *(const unsigned int*)&Bs[n * 40 + ((v >> 2) << 4) + (g << 3) + ((v & 3) << 1)];
      acc[0][tn] = wmma_f16(af[0].h, bf.h, acc[0][tn]);
      acc[1][tn] = wmma_f16(af[1].h, bf.h, acc[1][tn]);
    }
    __syncthreads();
  }

  // epilogue: C/D layout VGPR i -> m = i + 8*g, n = ln
#pragma unroll
  for (int tm = 0; tm < 2; tm++)
#pragma unroll
    for (int tn = 0; tn < 4; tn++)
#pragma unroll
      for (int i = 0; i < 8; i++) {
        int gr = row0 + wm * 32 + tm * 16 + i + g * 8;
        int gc = col0 + wn * 64 + tn * 16 + ln;
        float val = acc[tm][tn][i];
        if (MODE == 0) {
          int b = gr >> 11, n = gr & 2047;
          H16* dst = out0; int c = gc;
          if (gc >= split) { dst = out1; c = gc - split; }
          int h = c >> 6, d = c & 63;
          dst[(((size_t)(b * 16 + h)) * 2048 + n) * 64 + d] = (H16)val;
        } else {
          outF[(size_t)gr * Nmat + gc] = val + bias[gc];
        }
      }
}

// ---------------------------------------------------------------- flash attention
// Q,K,V in [b,h,n,d] f16; block = 128 q-rows of one (b,h); 8 waves x 16 rows.
// Online softmax over 64-key blocks; P round-trips through per-wave LDS.
__global__ __launch_bounds__(256) void attn128(
    const H16* __restrict__ Q, const H16* __restrict__ K,
    const H16* __restrict__ V, H16* __restrict__ Aout) {
  __shared__ H16 Vt[64 * 72];       // V^T tile: Vt[d][key], stride 72
  __shared__ H16 Pl[8 * 16 * 72];   // per-wave P tile: [row][key], stride 72
  const int t = threadIdx.x, lane = t & 31, wave = t >> 5;
  const int g = lane >> 4, ln = lane & 15;
  const int bh = blockIdx.y;
  const int b = bh >> 4, h = bh & 15;
  const int qbase = blockIdx.x * 128;
  const size_t hoff = (size_t)bh * 2048 * 64;
  const H16* Qp = Q + hoff;
  const H16* Kp = K + hoff;
  const H16* Vp = V + hoff;
  const int rowbase = qbase + wave * 16;

  // Q A-fragments (rows fixed for whole kernel), SCALE=1/8 folded in
  Frag16 qf[2];
#pragma unroll
  for (int f = 0; f < 2; f++)
#pragma unroll
    for (int v = 0; v < 8; v++) {
      int k0 = f * 32 + ((v >> 2) << 4) + (g << 3) + ((v & 3) << 1);
      qf[f].u[v] = *(const unsigned int*)(Qp + (size_t)(rowbase + ln) * 64 + k0);
    }
#pragma unroll
  for (int f = 0; f < 2; f++) qf[f].h *= (H16)0.125f;

  float rm[8], rs[8];
#pragma unroll
  for (int i = 0; i < 8; i++) { rm[i] = -3.0e38f; rs[i] = 0.f; }
  v8f Oacc[4] = {};

  const int vkp = t & 31, vdb = (t >> 5) * 8;
  H16* Pw = Pl + wave * (16 * 72);

  for (int kb = 0; kb < 2048; kb += 64) {
    __syncthreads();                    // all waves done with previous Vt
    // stage V^T (transpose; packed b32 DS stores)
    Half8 v0, v1;
    const H16* vp = Vp + (size_t)(kb + 2 * vkp) * 64 + vdb;
    v0.v = *(const float4*)vp;
    v1.v = *(const float4*)(vp + 64);
#pragma unroll
    for (int j = 0; j < 8; j++) {
      unsigned int lo = *(const unsigned short*)&v0.h[j];
      unsigned int hi = *(const unsigned short*)&v1.h[j];
      *(unsigned int*)&Vt[(vdb + j) * 72 + 2 * vkp] = lo | (hi << 16);
    }
    __syncthreads();

    // S = (Q*scale) @ K^T ; K row-major [key][d] is exactly the B-frag layout
    v8f S[4] = {};
#pragma unroll
    for (int kt = 0; kt < 4; kt++) {
      const H16* kp = Kp + (size_t)(kb + kt * 16 + ln) * 64;
#pragma unroll
      for (int f = 0; f < 2; f++) {
        Frag16 kf;
#pragma unroll
        for (int v = 0; v < 8; v++)
          kf.u[v] = *(const unsigned int*)(kp + f * 32 + ((v >> 2) << 4) + (g << 3) + ((v & 3) << 1));
        S[kt] = wmma_f16(qf[f].h, kf.h, S[kt]);
      }
    }

    // online softmax; row m = v + 8*g lives across a 16-lane group
#pragma unroll
    for (int v = 0; v < 8; v++) {
      float tmax = fmaxf(fmaxf(S[0][v], S[1][v]), fmaxf(S[2][v], S[3][v]));
      tmax = fmaxf(tmax, __shfl_xor(tmax, 1, 32));
      tmax = fmaxf(tmax, __shfl_xor(tmax, 2, 32));
      tmax = fmaxf(tmax, __shfl_xor(tmax, 4, 32));
      tmax = fmaxf(tmax, __shfl_xor(tmax, 8, 32));
      float nm = fmaxf(rm[v], tmax);
      float sc = __expf(rm[v] - nm);
      rm[v] = nm;
      rs[v] *= sc;
#pragma unroll
      for (int dt = 0; dt < 4; dt++) Oacc[dt][v] *= sc;
      float psum = 0.f;
#pragma unroll
      for (int kt = 0; kt < 4; kt++) {
        float p = __expf(S[kt][v] - nm);
        S[kt][v] = p;
        psum += p;
      }
      psum += __shfl_xor(psum, 1, 32);
      psum += __shfl_xor(psum, 2, 32);
      psum += __shfl_xor(psum, 4, 32);
      psum += __shfl_xor(psum, 8, 32);
      rs[v] += psum;
#pragma unroll
      for (int kt = 0; kt < 4; kt++)
        Pw[(v + g * 8) * 72 + kt * 16 + ln] = (H16)S[kt][v];
    }
    __syncthreads();                    // P visible, Vt still valid

    // O += P @ V : P A-frags from LDS, V B-frags from Vt[d][key]
#pragma unroll
    for (int f = 0; f < 2; f++) {
      Frag16 pf;
#pragma unroll
      for (int v = 0; v < 8; v++) {
        int k0 = f * 32 + ((v >> 2) << 4) + (g << 3) + ((v & 3) << 1);
        pf.u[v] = *(const unsigned int*)&Pw[ln * 72 + k0];
      }
#pragma unroll
      for (int dt = 0; dt < 4; dt++) {
        Frag16 vf;
#pragma unroll
        for (int v = 0; v < 8; v++)
          vf.u[v] = *(const unsigned int*)&Vt[(dt * 16 + ln) * 72 + f * 32 + ((v >> 2) << 4) + (g << 3) + ((v & 3) << 1)];
        Oacc[dt] = wmma_f16(pf.h, vf.h, Oacc[dt]);
      }
    }
  }

  // normalize and scatter to [b, n, h*64 + d] f16
#pragma unroll
  for (int i = 0; i < 8; i++) {
    float inv = 1.f / rs[i];
    int n = rowbase + i + g * 8;
#pragma unroll
    for (int dt = 0; dt < 4; dt++) {
      int c = h * 64 + dt * 16 + ln;
      Aout[((size_t)(b * 2048 + n)) * 1024 + c] = (H16)(Oacc[dt][i] * inv);
    }
  }
}

// ---------------------------------------------------------------- launch
extern "C" void kernel_launch(void* const* d_in, const int* in_sizes, int n_in,
                              void* d_out, int out_size, void* d_ws, size_t ws_size,
                              hipStream_t stream) {
  (void)in_sizes; (void)n_in; (void)out_size; (void)ws_size;
  const float* x_t  = (const float*)d_in[0];
  const float* x_s  = (const float*)d_in[1];
  const float* W_q  = (const float*)d_in[2];
  const float* W_kv = (const float*)d_in[3];
  const float* W_f  = (const float*)d_in[4];
  const float* b_f  = (const float*)d_in[5];
  float* out = (float*)d_out;

  const int NTOT = 4096, C = 1024;
  H16* ws    = (H16*)d_ws;
  H16* xt16  = ws;                       // 4096x1024
  H16* xs16  = xt16 + (size_t)NTOT * C;  // 4096x1024
  H16* wq16  = xs16 + (size_t)NTOT * C;  // 1024x1024
  H16* wkv16 = wq16 + (size_t)C * C;     // 1024x2048
  H16* wf16  = wkv16 + (size_t)C * 2 * C;// 1024x1024
  H16* q16   = wf16 + (size_t)C * C;     // [b,h,n,d]
  H16* k16   = q16 + (size_t)NTOT * C;
  H16* v16   = k16 + (size_t)NTOT * C;
  H16* a16   = v16 + (size_t)NTOT * C;   // [b,n,c]

  dim3 blk(256);
  cvt_f32_f16<<<(NTOT * C) / 2048, blk, 0, stream>>>(x_t, xt16, NTOT * C);
  cvt_f32_f16<<<(NTOT * C) / 2048, blk, 0, stream>>>(x_s, xs16, NTOT * C);
  cvt_f32_f16<<<(C * C) / 2048, blk, 0, stream>>>(W_q, wq16, C * C);
  cvt_f32_f16<<<(C * 2 * C) / 2048, blk, 0, stream>>>(W_kv, wkv16, C * 2 * C);
  cvt_f32_f16<<<(C * C) / 2048, blk, 0, stream>>>(W_f, wf16, C * C);

  // Q = x_t @ W_q -> head layout
  gemm128<0><<<dim3(1024 / 128, 4096 / 128), blk, 0, stream>>>(
      xt16, wq16, 4096, 1024, 1024, q16, nullptr, 1024, nullptr, nullptr);
  // K,V = x_s @ W_kv -> head layout (col split at 1024)
  gemm128<0><<<dim3(2048 / 128, 4096 / 128), blk, 0, stream>>>(
      xs16, wkv16, 4096, 2048, 1024, k16, v16, 1024, nullptr, nullptr);
  // flash attention
  attn128<<<dim3(2048 / 128, 32), blk, 0, stream>>>(q16, k16, v16, a16);
  // out = A @ W_fuse + b_fuse (f32)
  gemm128<1><<<dim3(1024 / 128, 4096 / 128), blk, 0, stream>>>(
      a16, wf16, 4096, 1024, 1024, nullptr, nullptr, 0, out, b_f);
}
